// ResidualAttentionBlockTensorCache_40200893890657
// MI455X (gfx1250) — matrix operand order repaired and checked by reading, not compiled, not run on hardware
//
#include <hip/hip_runtime.h>

// ---------------------------------------------------------------------------
// Whisper decoder residual attention block (B=8, TQ=8, D=1280, H=20, DH=64,
// TP=447, TC=1500, DFF=5120), fp32 in/out, WMMA f16 compute w/ f32 accumulate.
// Memory-bound workload (~250 MB streamed); tiles staged as packed f16 in LDS
// so fragment builds are contiguous ds_load_b128s feeding v_wmma.
// ---------------------------------------------------------------------------

typedef __attribute__((ext_vector_type(16))) _Float16 v16h;
typedef __attribute__((ext_vector_type(8)))  _Float16 v8h;
typedef __attribute__((ext_vector_type(4)))  _Float16 v4h;
typedef __attribute__((ext_vector_type(8)))  float    v8f;
typedef __attribute__((ext_vector_type(4)))  int      v4i;

#define FT_STRIDE 40   // f16 tile row stride: 80 bytes (16B multiple, 20-bank step)
#define KT_STRIDE 68   // f32 K/V tile stride: 272 bytes (16B multiple for b128 async)

// --- CDNA5 async global->LDS copy (ASYNCcnt), with portable fallback --------
#if __has_builtin(__builtin_amdgcn_global_load_async_to_lds_b128)
typedef __attribute__((address_space(1))) v4i* gas_v4i_ptr;   // global (AS1)
typedef __attribute__((address_space(3))) v4i* lds_v4i_ptr;   // LDS (AS3)
#define ASYNC_TILE_COPY(dst, src)                                              \
    __builtin_amdgcn_global_load_async_to_lds_b128(                            \
        (gas_v4i_ptr)(src), (lds_v4i_ptr)(dst), 0, 0)
#if __has_builtin(__builtin_amdgcn_s_wait_asynccnt)
#define ASYNC_WAIT() __builtin_amdgcn_s_wait_asynccnt(0)
#else
#define ASYNC_WAIT() asm volatile("s_wait_asynccnt 0" ::: "memory")
#endif
#else
#define ASYNC_TILE_COPY(dst, src) (*(float4*)(dst) = *(const float4*)(src))
#define ASYNC_WAIT()
#endif

// ---------------------------------------------------------------------------
// LayerNorm over rows of length 1280. grid = 64 rows, block = 256.
// ---------------------------------------------------------------------------
__global__ __launch_bounds__(256)
void ln_kernel(const float* __restrict__ x, const float* __restrict__ w,
               const float* __restrict__ bvec, float* __restrict__ out)
{
    __shared__ float s1[256], s2[256];
    const int row = blockIdx.x, tid = threadIdx.x;
    const float* xr = x + (size_t)row * 1280;
    float a = 0.f, b = 0.f;
    for (int i = tid; i < 1280; i += 256) { float v = xr[i]; a += v; b += v * v; }
    s1[tid] = a; s2[tid] = b;
    __syncthreads();
    for (int s = 128; s; s >>= 1) {
        if (tid < s) { s1[tid] += s1[tid + s]; s2[tid] += s2[tid + s]; }
        __syncthreads();
    }
    const float mean = s1[0] * (1.f / 1280.f);
    const float var  = s2[0] * (1.f / 1280.f) - mean * mean;
    const float rstd = rsqrtf(var + 1e-5f);
    float* orow = out + (size_t)row * 1280;
    for (int i = tid; i < 1280; i += 256)
        orow[i] = (xr[i] - mean) * rstd * w[i] + bvec[i];
}

// ---------------------------------------------------------------------------
// GEMM: out[64,N] = epilogue(A[64,K] @ W[K,N] (+bias) (gelu?) (+resid)).
// Block = 128 threads (4 waves), tile 64x64 output, K-chunks of 32.
// LDS tiles hold packed f16: At row-major [m][k], Wt transposed [col][k],
// so each lane's WMMA fragment is 2 contiguous ds_load_b128s.
// ---------------------------------------------------------------------------
__global__ __launch_bounds__(128)
void gemm64_wmma(const float* __restrict__ A, const float* __restrict__ W,
                 const float* __restrict__ bias, const float* __restrict__ resid,
                 float* __restrict__ out, int K, int N, int gelu)
{
    __shared__ _Float16 At[64 * FT_STRIDE];   // 64 rows x 32 k (padded)
    __shared__ _Float16 Wt[64 * FT_STRIDE];   // 64 cols x 32 k (padded)

    const int tid  = threadIdx.x;
    const int wave = tid >> 5;
    const int lane = tid & 31;
    const int half = lane >> 4;
    const int lq   = lane & 15;
    const int n0   = blockIdx.x * 64;
    const int col  = wave * 16 + lq;          // column within 64-wide tile

    v8f acc[4];
    for (int mt = 0; mt < 4; ++mt)
        for (int r = 0; r < 8; ++r) acc[mt][r] = 0.f;

    // W staging: this thread owns one 4x4 (k x col) block of the 32x64 chunk.
    const int kb = (tid & 7) * 4;
    const int cb = (tid >> 3) * 4;

    for (int k0 = 0; k0 < K; k0 += 32) {
        // ---- stage A chunk (64x32), convert to f16 at store time ----
        for (int j = 0; j < 4; ++j) {
            int q   = tid + j * 128;          // 512 float4 total
            int row = q >> 3;
            int c4  = (q & 7) * 4;
            const float4 v = *(const float4*)(A + (size_t)row * K + k0 + c4);
            v4h hv = { (_Float16)v.x, (_Float16)v.y, (_Float16)v.z, (_Float16)v.w };
            *(v4h*)&At[row * FT_STRIDE + c4] = hv;
        }
        // ---- stage W chunk (32x64) transposed via 4x4 register blocks ----
        {
            float r0[4], r1[4], r2[4], r3[4];
            const float* wp = W + (size_t)(k0 + kb) * N + n0 + cb;
            float4 t;
            t = *(const float4*)(wp);                 r0[0]=t.x; r0[1]=t.y; r0[2]=t.z; r0[3]=t.w;
            t = *(const float4*)(wp + (size_t)N);     r1[0]=t.x; r1[1]=t.y; r1[2]=t.z; r1[3]=t.w;
            t = *(const float4*)(wp + (size_t)2*N);   r2[0]=t.x; r2[1]=t.y; r2[2]=t.z; r2[3]=t.w;
            t = *(const float4*)(wp + (size_t)3*N);   r3[0]=t.x; r3[1]=t.y; r3[2]=t.z; r3[3]=t.w;
            for (int c = 0; c < 4; ++c) {
                v4h hv = { (_Float16)r0[c], (_Float16)r1[c], (_Float16)r2[c], (_Float16)r3[c] };
                *(v4h*)&Wt[(cb + c) * FT_STRIDE + kb] = hv;
            }
        }
        // CDNA5 global_prefetch of next chunk to hide HBM latency.
        if (k0 + 32 < K) {
            __builtin_prefetch(W + (size_t)(k0 + 32 + kb) * N + n0 + cb, 0, 1);
            __builtin_prefetch(A + (size_t)((tid >> 1) & 63) * K + k0 + 32, 0, 1);
        }
        __syncthreads();

        // ---- fragments: contiguous packed-f16 runs per documented layout ----
        // B (32x16): elements 0..7 <-> K = half*8..+7; 8..15 <-> 16+half*8..+7
        const _Float16* bp = &Wt[col * FT_STRIDE];
        v8h b0 = *(const v8h*)(bp + half * 8);
        v8h b1 = *(const v8h*)(bp + 16 + half * 8);
        v16h bfrag = __builtin_shufflevector(b0, b1,
            0,1,2,3,4,5,6,7,8,9,10,11,12,13,14,15);

        for (int mt = 0; mt < 4; ++mt) {
            // A (16x32): elements 0..7 <-> K = half*8..+7; 8..15 <-> 16+half*8..+7
            const _Float16* ap = &At[(mt * 16 + lq) * FT_STRIDE];
            v8h a0 = *(const v8h*)(ap + half * 8);
            v8h a1 = *(const v8h*)(ap + 16 + half * 8);
            v16h afrag = __builtin_shufflevector(a0, a1,
                0,1,2,3,4,5,6,7,8,9,10,11,12,13,14,15);
            acc[mt] = __builtin_amdgcn_wmma_f32_16x16x32_f16(
                false, afrag, false, bfrag, (short)0, acc[mt], false, false);
        }
        __syncthreads();
    }

    // ---- epilogue: bias -> (gelu) -> (+resid) -> store ----
    const float bval = bias ? bias[n0 + col] : 0.f;
    for (int mt = 0; mt < 4; ++mt) {
        for (int r = 0; r < 8; ++r) {
            const int m = mt * 16 + half * 8 + r;   // C/D layout: M = half*8 + r
            float val = acc[mt][r] + bval;
            if (gelu) val = 0.5f * val * (1.f + erff(val * 0.70710678118f));
            if (resid) val += resid[(size_t)m * N + n0 + col];
            out[(size_t)m * N + n0 + col] = val;
        }
    }
}

// ---------------------------------------------------------------------------
// Attention for one (batch, head): q[8,64] vs K/V with Tk keys.
// Keys/values come from [cache rows 0..Tp) then [new rows Tp..Tk).
// grid = B*H = 160 blocks, block = 256 threads (8 waves).
// K/V tiles staged with CDNA5 async global->LDS b128 copies (ASYNCcnt).
// Logits kept in dynamic LDS (8 x Tk); wave-per-row softmax via shfl_xor.
// ---------------------------------------------------------------------------
__global__ __launch_bounds__(256)
void attn_kernel(const float* __restrict__ q,
                 const float* __restrict__ kc, const float* __restrict__ kn,
                 const float* __restrict__ vc, const float* __restrict__ vn,
                 const float* __restrict__ mask,
                 float* __restrict__ ctx,
                 int Tp, int Tk)
{
    extern __shared__ float smem[];
    float* qs = smem;                 // 8*64
    float* kt = smem + 512;           // 32*KT_STRIDE (reused for V tiles)
    float* ls = kt + 32 * KT_STRIDE;  // 8*Tk logits/weights

    const int tid  = threadIdx.x;
    const int b    = blockIdx.x / 20;
    const int h    = blockIdx.x % 20;
    const size_t hoff = (size_t)h * 64;
    const float sc = 0.125f;          // (DH^-0.25)^2 applied to q

    for (int i = tid; i < 512; i += 256) {
        int qi = i >> 6, d = i & 63;
        qs[i] = q[(size_t)(b * 8 + qi) * 1280 + hoff + d] * sc;
    }
    __syncthreads();

    // ---- logits = q . k (+ mask) ----
    for (int t0 = 0; t0 < Tk; t0 += 32) {
        for (int j = 0; j < 2; ++j) {              // 512 b128 payloads per tile
            int f  = j * 256 + tid;
            int kk = f >> 4;                       // key row in tile
            int c4 = (f & 15) * 4;                 // 4-float column base
            int t  = t0 + kk;
            if (t < Tk) {
                const float* src = (t < Tp)
                    ? (kc + (size_t)(b * Tp + t) * 1280 + hoff + c4)
                    : (kn + (size_t)(b * 8 + (t - Tp)) * 1280 + hoff + c4);
                ASYNC_TILE_COPY(&kt[kk * KT_STRIDE + c4], src);
            }
        }
        ASYNC_WAIT();
        __syncthreads();
        {
            int kk = tid >> 3, qi = tid & 7;       // 32 keys x 8 queries
            int t  = t0 + kk;
            if (t < Tk) {
                const float* qr = qs + qi * 64;
                const float* kr = kt + kk * KT_STRIDE;
                float dot = 0.f;
                for (int d = 0; d < 64; ++d) dot += qr[d] * kr[d];
                if (mask) dot += mask[qi * Tk + t];
                ls[qi * Tk + t] = dot;
            }
        }
        __syncthreads();
    }

    // ---- softmax, one wave32 per query row ----
    {
        const int row = tid >> 5, lane = tid & 31;
        float m = -3.4e38f;
        for (int t = lane; t < Tk; t += 32) m = fmaxf(m, ls[row * Tk + t]);
        for (int off = 16; off; off >>= 1) m = fmaxf(m, __shfl_xor(m, off, 32));
        float s = 0.f;
        for (int t = lane; t < Tk; t += 32) {
            float e = __expf(ls[row * Tk + t] - m);
            ls[row * Tk + t] = e;
            s += e;
        }
        for (int off = 16; off; off >>= 1) s += __shfl_xor(s, off, 32);
        const float inv = 1.f / s;
        for (int t = lane; t < Tk; t += 32) ls[row * Tk + t] *= inv;
    }
    __syncthreads();

    // ---- ctx = weights @ V ----  thread -> (query row, 2 head dims)
    const int qi = tid >> 5;
    const int d0 = (tid & 31) * 2;
    float a0 = 0.f, a1 = 0.f;
    for (int t0 = 0; t0 < Tk; t0 += 32) {
        for (int j = 0; j < 2; ++j) {              // stage 32x64 value tile
            int f  = j * 256 + tid;
            int kk = f >> 4;
            int c4 = (f & 15) * 4;
            int t  = t0 + kk;
            if (t < Tk) {
                const float* src = (t < Tp)
                    ? (vc + (size_t)(b * Tp + t) * 1280 + hoff + c4)
                    : (vn + (size_t)(b * 8 + (t - Tp)) * 1280 + hoff + c4);
                ASYNC_TILE_COPY(&kt[kk * KT_STRIDE + c4], src);
            }
        }
        ASYNC_WAIT();
        __syncthreads();
        const int lim = min(32, Tk - t0);
        for (int kk = 0; kk < lim; ++kk) {
            const float w = ls[qi * Tk + t0 + kk];
            a0 += w * kt[kk * KT_STRIDE + d0];
            a1 += w * kt[kk * KT_STRIDE + d0 + 1];
        }
        __syncthreads();
    }
    ctx[(size_t)(b * 8 + qi) * 1280 + hoff + d0]     = a0;
    ctx[(size_t)(b * 8 + qi) * 1280 + hoff + d0 + 1] = a1;
}

// ---------------------------------------------------------------------------
// Host-side launch sequence (graph-capture safe: stream ops only).
// ---------------------------------------------------------------------------
extern "C" void kernel_launch(void* const* d_in, const int* in_sizes, int n_in,
                              void* d_out, int out_size, void* d_ws, size_t ws_size,
                              hipStream_t stream)
{
    (void)in_sizes; (void)n_in; (void)out_size; (void)ws_size;

    const float* x    = (const float*)d_in[0];
    const float* skc  = (const float*)d_in[1];
    const float* svc  = (const float*)d_in[2];
    const float* ck   = (const float*)d_in[3];
    const float* cv   = (const float*)d_in[4];
    const float* mask = (const float*)d_in[5];
    const float* wq   = (const float*)d_in[6];
    const float* bq   = (const float*)d_in[7];
    const float* wk   = (const float*)d_in[8];
    const float* wv   = (const float*)d_in[9];
    const float* bv   = (const float*)d_in[10];
    const float* wo   = (const float*)d_in[11];
    const float* bo   = (const float*)d_in[12];
    const float* cwq  = (const float*)d_in[13];
    const float* cbq  = (const float*)d_in[14];
    const float* cwo  = (const float*)d_in[15];
    const float* cbo  = (const float*)d_in[16];
    const float* ln1w = (const float*)d_in[17];
    const float* ln1b = (const float*)d_in[18];
    const float* ln2w = (const float*)d_in[19];
    const float* ln2b = (const float*)d_in[20];
    const float* ln3w = (const float*)d_in[21];
    const float* ln3b = (const float*)d_in[22];
    const float* w1   = (const float*)d_in[23];
    const float* b1   = (const float*)d_in[24];
    const float* w2   = (const float*)d_in[25];
    const float* b2   = (const float*)d_in[26];

    const int RC = 64 * 1280;            // one 64x1280 activation buffer

    float* outx = (float*)d_out;         // x_out
    float* k1   = outx + RC;             // k1 output (also self-attn input)
    float* v1   = outx + 2 * RC;         // v1 output

    float* ws   = (float*)d_ws;
    float* xl   = ws;                    // LN output (reused)
    float* qbuf = ws + (size_t)RC;       // q / qc
    float* ctx  = ws + (size_t)2 * RC;   // attention context (reused)
    float* x1   = ws + (size_t)3 * RC;   // after self-attn residual
    float* x2   = ws + (size_t)4 * RC;   // after cross-attn residual
    float* hbuf = ws + (size_t)5 * RC;   // MLP hidden 64x5120

    const size_t smem_self  = (512 + 32 * KT_STRIDE + 8 * 455)  * sizeof(float);
    const size_t smem_cross = (512 + 32 * KT_STRIDE + 8 * 1500) * sizeof(float);

    // --- self-attention with KV cache ---
    ln_kernel<<<64, 256, 0, stream>>>(x, ln1w, ln1b, xl);
    gemm64_wmma<<<20, 128, 0, stream>>>(xl, wq, bq, nullptr, qbuf, 1280, 1280, 0);
    gemm64_wmma<<<20, 128, 0, stream>>>(xl, wk, nullptr, nullptr, k1, 1280, 1280, 0);
    gemm64_wmma<<<20, 128, 0, stream>>>(xl, wv, bv, nullptr, v1, 1280, 1280, 0);
    attn_kernel<<<160, 256, smem_self, stream>>>(qbuf, skc, k1, svc, v1, mask, ctx, 447, 455);
    gemm64_wmma<<<20, 128, 0, stream>>>(ctx, wo, bo, x, x1, 1280, 1280, 0);

    // --- cross-attention ---
    ln_kernel<<<64, 256, 0, stream>>>(x1, ln2w, ln2b, xl);
    gemm64_wmma<<<20, 128, 0, stream>>>(xl, cwq, cbq, nullptr, qbuf, 1280, 1280, 0);
    attn_kernel<<<160, 256, smem_cross, stream>>>(qbuf, ck, nullptr, cv, nullptr, nullptr, ctx, 1500, 1500);
    gemm64_wmma<<<20, 128, 0, stream>>>(ctx, cwo, cbo, x1, x2, 1280, 1280, 0);

    // --- MLP ---
    ln_kernel<<<64, 256, 0, stream>>>(x2, ln3w, ln3b, xl);
    gemm64_wmma<<<80, 128, 0, stream>>>(xl, w1, b1, nullptr, hbuf, 1280, 5120, 1);
    gemm64_wmma<<<20, 128, 0, stream>>>(hbuf, w2, b2, x2, outx, 5120, 1280, 0);
}
